// SimpleDecoder_10934986736070
// MI455X (gfx1250) — matrix operand order repaired and checked by reading, not compiled, not run on hardware
//
#include <hip/hip_runtime.h>
#include <hip/hip_bf16.h>

// ---------------------------------------------------------------------------
// LSTM greedy decoder for gfx1250 (MI455X).
//  - All GEMMs via v_wmma_f32_16x16x32_f16 (f16 inputs, f32 accumulate).
//  - Weights converted f32->f16 once per launch into d_ws; W_fc_f16 (32.8MB)
//    is L2-resident (192MB L2) so the per-step weight stream never touches HBM
//    after the first step.
//  - Sequential T=128 loop driven from host; every kernel on `stream`.
// ---------------------------------------------------------------------------

typedef __attribute__((ext_vector_type(16))) _Float16 v16h;
typedef __attribute__((ext_vector_type(8)))  _Float16 v8h;
typedef __attribute__((ext_vector_type(8)))  float    v8f;

#define VOCAB  32000
#define EMBED  256
#define HIDDEN 512
#define BATCH  32
#define MAXLEN 128
#define GATES  (4 * HIDDEN)   // 2048

// ---------------------------------------------------------------------------
// WMMA helpers (wave32; 16x16x32 f16 -> f32)
// ---------------------------------------------------------------------------
__device__ __forceinline__ v8f wmma_f16(v16h a, v16h b, v8f c) {
  return __builtin_amdgcn_wmma_f32_16x16x32_f16(
      /*neg_a=*/false, a, /*neg_b=*/false, b,
      /*c_mod=*/(short)0, c, /*reuse_a=*/false, /*reuse_b=*/false);
}

// A-matrix fragment: src is row-major [rows x ld] f16, tile at (row0, k0).
// ISA layout (16-bit A 16x32): lanes 0-15 -> M=lane, K={0..7} in v0..3,
// K={16..23} in v4..7; lanes 16-31 -> M=lane-16, K={8..15},{24..31}.
__device__ __forceinline__ v16h load_frag_a(const _Float16* __restrict__ src,
                                            int ld, int row0, int k0, int lane) {
  int half = lane >> 4;
  int row  = row0 + (lane & 15);
  const _Float16* p = src + (size_t)row * ld + k0 + half * 8;
  v8h lo = *(const v8h*)(p);        // K = k0 + half*8      .. +7
  v8h hi = *(const v8h*)(p + 16);   // K = k0 + 16 + half*8 .. +7
  return __builtin_shufflevector(lo, hi, 0, 1, 2, 3, 4, 5, 6, 7,
                                 8, 9, 10, 11, 12, 13, 14, 15);
}

// B-matrix fragment: element (k, n) = W[n][k] with W row-major [N x ld].
// ISA layout (16-bit B 32x16): lanes 0-15 -> N=lane, K={0..7},{16..23};
// lanes 16-31 -> N=lane-16, K={8..15},{24..31}.
__device__ __forceinline__ v16h load_frag_b(const _Float16* __restrict__ W,
                                            int ld, int n0, int k0, int lane) {
  int half = lane >> 4;
  int n = n0 + (lane & 15);
  const _Float16* p = W + (size_t)n * ld + k0 + half * 8;
  v8h lo = *(const v8h*)(p);
  v8h hi = *(const v8h*)(p + 16);
  return __builtin_shufflevector(lo, hi, 0, 1, 2, 3, 4, 5, 6, 7,
                                 8, 9, 10, 11, 12, 13, 14, 15);
}

// ---------------------------------------------------------------------------
// Prep kernels
// ---------------------------------------------------------------------------
__global__ void convert_f32_to_f16(const float* __restrict__ src,
                                   _Float16* __restrict__ dst, int n) {
  int stride = gridDim.x * blockDim.x;
  for (int i = blockIdx.x * blockDim.x + threadIdx.x; i < n; i += stride)
    dst[i] = (_Float16)src[i];
}

__global__ void init_state(float* __restrict__ h, float* __restrict__ c,
                           _Float16* __restrict__ h16, int* __restrict__ tok) {
  int i = blockIdx.x * blockDim.x + threadIdx.x;   // 16384 threads
  h[i] = 0.0f;
  c[i] = 0.0f;
  h16[i] = (_Float16)0.0f;
  if (i < BATCH) tok[i] = 0;                       // argmax(zeros) == 0
}

// x = embed[tok]  -> f16   (32 x 256)
__global__ void gather_x(const int* __restrict__ tok,
                         const float* __restrict__ embed,
                         _Float16* __restrict__ x16) {
  int i = blockIdx.x * blockDim.x + threadIdx.x;   // 8192 threads
  int b = i >> 8;
  int e = i & 255;
  x16[i] = (_Float16)embed[(size_t)tok[b] * EMBED + e];
}

// ---------------------------------------------------------------------------
// gates[32 x 2048] = x16[32x256] @ Wih16^T + h16[32x512] @ Whh16^T
// grid: 16 blocks x 256 threads (8 waves); wave -> one N-tile, both M-tiles.
// ---------------------------------------------------------------------------
__global__ void gates_gemm(const _Float16* __restrict__ x16,
                           const _Float16* __restrict__ h16,
                           const _Float16* __restrict__ Wih16,
                           const _Float16* __restrict__ Whh16,
                           float* __restrict__ gates) {
  int lane = threadIdx.x & 31;
  int wave = threadIdx.x >> 5;
  int n0 = (blockIdx.x * 8 + wave) * 16;   // 0..2032

  v8f acc0 = {};   // rows 0..15
  v8f acc1 = {};   // rows 16..31
  for (int k0 = 0; k0 < EMBED; k0 += 32) {
    v16h bf = load_frag_b(Wih16, EMBED, n0, k0, lane);
    v16h a0 = load_frag_a(x16, EMBED, 0, k0, lane);
    v16h a1 = load_frag_a(x16, EMBED, 16, k0, lane);
    acc0 = wmma_f16(a0, bf, acc0);
    acc1 = wmma_f16(a1, bf, acc1);
  }
  for (int k0 = 0; k0 < HIDDEN; k0 += 32) {
    v16h bf = load_frag_b(Whh16, HIDDEN, n0, k0, lane);
    v16h a0 = load_frag_a(h16, HIDDEN, 0, k0, lane);
    v16h a1 = load_frag_a(h16, HIDDEN, 16, k0, lane);
    acc0 = wmma_f16(a0, bf, acc0);
    acc1 = wmma_f16(a1, bf, acc1);
  }

  // D layout: lane -> column n0+(lane&15); vgpr r -> row r + 8*(lane>>4)
  int half = lane >> 4;
  int n = n0 + (lane & 15);
  for (int r = 0; r < 8; ++r) {
    gates[(size_t)(r + 8 * half) * GATES + n]      = acc0[r];
    gates[(size_t)(16 + r + 8 * half) * GATES + n] = acc1[r];
  }
}

// ---------------------------------------------------------------------------
// LSTM pointwise update (PyTorch gate order i,f,g,o). 16384 threads.
// ---------------------------------------------------------------------------
__global__ void lstm_cell(const float* __restrict__ gates,
                          const float* __restrict__ b_ih,
                          const float* __restrict__ b_hh,
                          float* __restrict__ h, float* __restrict__ c,
                          _Float16* __restrict__ h16) {
  int j = blockIdx.x * blockDim.x + threadIdx.x;   // 0..16383
  int b = j >> 9;
  int u = j & 511;
  const float* g = gates + (size_t)b * GATES;
  float gi = g[u]        + b_ih[u]        + b_hh[u];
  float gf = g[512 + u]  + b_ih[512 + u]  + b_hh[512 + u];
  float gg = g[1024 + u] + b_ih[1024 + u] + b_hh[1024 + u];
  float go = g[1536 + u] + b_ih[1536 + u] + b_hh[1536 + u];
  float i_ = 1.0f / (1.0f + expf(-gi));
  float f_ = 1.0f / (1.0f + expf(-gf));
  float g_ = tanhf(gg);
  float o_ = 1.0f / (1.0f + expf(-go));
  float cn = f_ * c[j] + i_ * g_;
  float hn = o_ * tanhf(cn);
  c[j] = cn;
  h[j] = hn;
  h16[j] = (_Float16)hn;
}

// ---------------------------------------------------------------------------
// logits[32 x 32000] = h16[32x512] @ Wfc16^T + b_fc  -> d_out[:, t, :]
// grid: 250 blocks x 256 threads = 2000 waves = 2000 N-tiles; each wave
// computes both M-tiles so each W_fc fragment is fetched exactly once/step.
// ---------------------------------------------------------------------------
__global__ void logits_gemm(const _Float16* __restrict__ h16,
                            const _Float16* __restrict__ Wfc16,
                            const float* __restrict__ b_fc,
                            float* __restrict__ out, int t) {
  int lane = threadIdx.x & 31;
  int wave = threadIdx.x >> 5;
  int n0 = (blockIdx.x * 8 + wave) * 16;   // 0..31984

  v8f acc0 = {};
  v8f acc1 = {};
  for (int k0 = 0; k0 < HIDDEN; k0 += 32) {
    v16h bf = load_frag_b(Wfc16, HIDDEN, n0, k0, lane);
    v16h a0 = load_frag_a(h16, HIDDEN, 0, k0, lane);
    v16h a1 = load_frag_a(h16, HIDDEN, 16, k0, lane);
    acc0 = wmma_f16(a0, bf, acc0);
    acc1 = wmma_f16(a1, bf, acc1);
  }

  int half = lane >> 4;
  int n = n0 + (lane & 15);
  float bias = b_fc[n];
  size_t tv = (size_t)t * VOCAB + n;
  const size_t rowstride = (size_t)MAXLEN * VOCAB;
  for (int r = 0; r < 8; ++r) {
    int m = r + 8 * half;
    out[(size_t)m * rowstride + tv]        = acc0[r] + bias;
    out[(size_t)(16 + m) * rowstride + tv] = acc1[r] + bias;
  }
}

// ---------------------------------------------------------------------------
// tok[b] = argmax_v logits[b, t, v]  (first-max tie-break like jnp.argmax)
// ---------------------------------------------------------------------------
__global__ void argmax_row(const float* __restrict__ out, int t,
                           int* __restrict__ tok) {
  int b = blockIdx.x;
  const float* row = out + (size_t)b * MAXLEN * VOCAB + (size_t)t * VOCAB;
  __shared__ float sv[256];
  __shared__ int   si[256];
  float best = -__builtin_huge_valf();
  int bi = 0;
  for (int v = threadIdx.x; v < VOCAB; v += 256) {
    float x = row[v];
    if (x > best) { best = x; bi = v; }
  }
  sv[threadIdx.x] = best;
  si[threadIdx.x] = bi;
  __syncthreads();
  for (int s = 128; s > 0; s >>= 1) {
    if (threadIdx.x < s) {
      float v2 = sv[threadIdx.x + s];
      int   i2 = si[threadIdx.x + s];
      if (v2 > sv[threadIdx.x] ||
          (v2 == sv[threadIdx.x] && i2 < si[threadIdx.x])) {
        sv[threadIdx.x] = v2;
        si[threadIdx.x] = i2;
      }
    }
    __syncthreads();
  }
  if (threadIdx.x == 0) tok[b] = si[0];
}

// ---------------------------------------------------------------------------
// Host side
// ---------------------------------------------------------------------------
static inline size_t align_up(size_t x, size_t a) { return (x + a - 1) & ~(a - 1); }

extern "C" void kernel_launch(void* const* d_in, const int* in_sizes, int n_in,
                              void* d_out, int out_size, void* d_ws, size_t ws_size,
                              hipStream_t stream) {
  (void)in_sizes; (void)n_in; (void)out_size; (void)ws_size;

  // setup_inputs order: text, embed, W_ih, W_hh, b_ih, b_hh, W_fc, b_fc
  const float* embed = (const float*)d_in[1];
  const float* W_ih  = (const float*)d_in[2];
  const float* W_hh  = (const float*)d_in[3];
  const float* b_ih  = (const float*)d_in[4];
  const float* b_hh  = (const float*)d_in[5];
  const float* W_fc  = (const float*)d_in[6];
  const float* b_fc  = (const float*)d_in[7];
  float* out = (float*)d_out;

  // Workspace carve-up (all offsets 256B aligned).
  char* ws = (char*)d_ws;
  size_t off = 0;
  _Float16* Wfc16 = (_Float16*)(ws + off); off = align_up(off + (size_t)VOCAB * HIDDEN * 2, 256);
  _Float16* Wih16 = (_Float16*)(ws + off); off = align_up(off + (size_t)GATES * EMBED * 2, 256);
  _Float16* Whh16 = (_Float16*)(ws + off); off = align_up(off + (size_t)GATES * HIDDEN * 2, 256);
  float*    h     = (float*)(ws + off);    off = align_up(off + (size_t)BATCH * HIDDEN * 4, 256);
  float*    c     = (float*)(ws + off);    off = align_up(off + (size_t)BATCH * HIDDEN * 4, 256);
  _Float16* h16   = (_Float16*)(ws + off); off = align_up(off + (size_t)BATCH * HIDDEN * 2, 256);
  _Float16* x16   = (_Float16*)(ws + off); off = align_up(off + (size_t)BATCH * EMBED * 2, 256);
  float*    gates = (float*)(ws + off);    off = align_up(off + (size_t)BATCH * GATES * 4, 256);
  int*      tok   = (int*)(ws + off);      off = align_up(off + (size_t)BATCH * 4, 256);

  // One-time per launch: f32 -> f16 weight copies (deterministic, recomputed).
  convert_f32_to_f16<<<2048, 256, 0, stream>>>(W_fc, Wfc16, VOCAB * HIDDEN);
  convert_f32_to_f16<<<1024, 256, 0, stream>>>(W_ih, Wih16, GATES * EMBED);
  convert_f32_to_f16<<<1024, 256, 0, stream>>>(W_hh, Whh16, GATES * HIDDEN);
  init_state<<<64, 256, 0, stream>>>(h, c, h16, tok);

  // Sequential greedy decode.
  for (int t = 0; t < MAXLEN; ++t) {
    gather_x<<<32, 256, 0, stream>>>(tok, embed, x16);
    gates_gemm<<<16, 256, 0, stream>>>(x16, h16, Wih16, Whh16, gates);
    lstm_cell<<<64, 256, 0, stream>>>(gates, b_ih, b_hh, h, c, h16);
    logits_gemm<<<250, 256, 0, stream>>>(h16, Wfc16, b_fc, out, t);
    argmax_row<<<32, 256, 0, stream>>>(out, t, tok);
  }
}
